// Regressor_73289321939414
// MI455X (gfx1250) — compile-verified, matrix operand order checked by reading, not run on hardware
//
#include <hip/hip_runtime.h>

// ---------------------------------------------------------------- types ----
typedef __attribute__((ext_vector_type(16))) __bf16 v16bf;
typedef __attribute__((ext_vector_type(8)))  __bf16 v8bf;
typedef __attribute__((ext_vector_type(8)))  float  v8f;

// ---------------------------------------------------------- problem dims ---
constexpr int BATCH = 256;
constexpr int NNODE = 512;
constexpr int NEDGE = 8192;
constexpr int FIN   = 128;
constexpr int NHID  = 256;
constexpr int CATC  = 896;            // FIN + 3*NHID
constexpr int MROWS = BATCH * NNODE;  // 131072

// ------------------------------------------------------------ WMMA GEMM ----
// Tiles: 128x128x32 per workgroup, 256 threads = 8 waves, each wave 32x64.
#define BM 128
#define BN 128
#define BK 32
#define LDSP 40   // padded row stride in bf16 elems (80B, 16B-aligned)

__device__ __forceinline__ v8bf pack8(const float* __restrict__ f) {
  v8bf r = {(__bf16)f[0], (__bf16)f[1], (__bf16)f[2], (__bf16)f[3],
            (__bf16)f[4], (__bf16)f[5], (__bf16)f[6], (__bf16)f[7]};
  return r;
}

// Stage one 128x32 fp32->bf16 A tile: thread t loads 16 contiguous floats of
// row t/2 as 4x float4, stores 2x 16B chunks into LDS.
__device__ __forceinline__ void stage_a(const float* __restrict__ A, size_t lda,
                                        size_t blockM, int k0, int tid,
                                        __bf16 (*As)[LDSP]) {
  const int arow  = tid >> 1;
  const int acol0 = (tid & 1) * 16;
  const float* ap = A + (blockM + arow) * lda + k0 + acol0;
  float4 q0 = ((const float4*)ap)[0];
  float4 q1 = ((const float4*)ap)[1];
  float4 q2 = ((const float4*)ap)[2];
  float4 q3 = ((const float4*)ap)[3];
  float lo[8] = {q0.x, q0.y, q0.z, q0.w, q1.x, q1.y, q1.z, q1.w};
  float hi[8] = {q2.x, q2.y, q2.z, q2.w, q3.x, q3.y, q3.z, q3.w};
  *(v8bf*)&As[arow][acol0]     = pack8(lo);
  *(v8bf*)&As[arow][acol0 + 8] = pack8(hi);
}

// Stage one 32x128 fp32 weight tile TRANSPOSED into Bs[n][k] (bf16): thread t
// owns column n = t&127, k-half = 16*(t>>7); 16 strided global reads that are
// coalesced across lanes, 2x 16B LDS stores.
__device__ __forceinline__ void stage_b(const float* __restrict__ Bw, int ldb,
                                        int blockN, int k0, int tid,
                                        __bf16 (*Bs)[LDSP]) {
  const int bn  = tid & 127;
  const int bkh = (tid >> 7) * 16;
  const float* bp = Bw + (size_t)(k0 + bkh) * ldb + blockN + bn;
  float t[16];
#pragma unroll
  for (int i = 0; i < 16; ++i) t[i] = bp[(size_t)i * ldb];
  *(v8bf*)&Bs[bn][bkh]     = pack8(t);
  *(v8bf*)&Bs[bn][bkh + 8] = pack8(t + 8);
}

// ISA 7.12.2 fragment load: lane<16 holds K=0..7,16..23; lane>=16 K=8..15,24..31
__device__ __forceinline__ v16bf load_frag(const __bf16 (*S)[LDSP], int row, int koff) {
  v8bf lo = *(const v8bf*)&S[row][koff];
  v8bf hi = *(const v8bf*)&S[row][koff + 16];
  return __builtin_shufflevector(lo, hi, 0, 1, 2, 3, 4, 5, 6, 7, 8, 9, 10, 11,
                                 12, 13, 14, 15);
}

template <int ACT>  // 0=none 1=tanh 2=sigmoid
__global__ __launch_bounds__(256) void wmma_gemm_kernel(
    const float* __restrict__ A, int lda,
    const float* __restrict__ Bw, int ldb,
    const float* __restrict__ bias,
    float* __restrict__ C, int ldc, int K) {
  __shared__ __bf16 As[BM][LDSP];
  __shared__ __bf16 Bs[BN][LDSP];

  const int tid   = threadIdx.x;
  const int lane  = tid & 31;
  const int wid   = tid >> 5;
  const int waveM = wid >> 1;   // 0..3
  const int waveN = wid & 1;    // 0..1

  const size_t blockM = (size_t)blockIdx.x * BM;
  const int    blockN = blockIdx.y * BN;

  v8f acc[2][4];
#pragma unroll
  for (int i = 0; i < 2; ++i)
#pragma unroll
    for (int j = 0; j < 4; ++j) acc[i][j] = (v8f)(0.0f);

  const int rsel = lane & 15;
  const int koff = (lane >> 4) * 8;

  for (int k0 = 0; k0 < K; k0 += BK) {
    __syncthreads();
    stage_a(A, (size_t)lda, blockM, k0, tid, As);
    stage_b(Bw, ldb, blockN, k0, tid, Bs);
    if (k0 + BK < K)  // gfx1250 global_prefetch of next A k-tile
      __builtin_prefetch(A + (blockM + (tid >> 1)) * (size_t)lda + k0 + BK, 0, 0);
    __syncthreads();

    v16bf af[2], bfr[4];
#pragma unroll
    for (int mt = 0; mt < 2; ++mt)
      af[mt] = load_frag(As, waveM * 32 + mt * 16 + rsel, koff);
#pragma unroll
    for (int nt = 0; nt < 4; ++nt)
      bfr[nt] = load_frag(Bs, waveN * 64 + nt * 16 + rsel, koff);
#pragma unroll
    for (int mt = 0; mt < 2; ++mt)
#pragma unroll
      for (int nt = 0; nt < 4; ++nt)
        acc[mt][nt] = __builtin_amdgcn_wmma_f32_16x16x32_bf16(
            false, af[mt], false, bfr[nt], (short)0, acc[mt][nt], false, false);
  }

  const int mlane = (lane >> 4) << 3;
  const int nlane = lane & 15;
#pragma unroll
  for (int mt = 0; mt < 2; ++mt)
#pragma unroll
    for (int nt = 0; nt < 4; ++nt)
#pragma unroll
      for (int r = 0; r < 8; ++r) {
        const size_t m = blockM + waveM * 32 + mt * 16 + r + mlane;
        const int n = blockN + waveN * 64 + nt * 16 + nlane;
        float v = acc[mt][nt][r];
        if (bias) v += bias[n];
        if (ACT == 1) v = tanhf(v);
        if (ACT == 2) v = 1.0f / (1.0f + expf(-v));
        C[m * (size_t)ldc + n] = v;
      }
}

// Dual-weight readout GEMM: stages the A (cat) tile once, runs it against BOTH
// the sigmoid and tanh weight matrices -> halves HBM reads of the 470MB cat.
__global__ __launch_bounds__(256) void wmma_gemm_dual_kernel(
    const float* __restrict__ A, int lda,
    const float* __restrict__ Ws, const float* __restrict__ Wt, int ldb,
    const float* __restrict__ bs, const float* __restrict__ bt,
    float* __restrict__ S, float* __restrict__ T, int ldc, int K) {
  __shared__ __bf16 As[BM][LDSP];
  __shared__ __bf16 BsS[BN][LDSP];
  __shared__ __bf16 BsT[BN][LDSP];

  const int tid   = threadIdx.x;
  const int lane  = tid & 31;
  const int wid   = tid >> 5;
  const int waveM = wid >> 1;
  const int waveN = wid & 1;

  const size_t blockM = (size_t)blockIdx.x * BM;
  const int    blockN = blockIdx.y * BN;

  v8f aS[2][4], aT[2][4];
#pragma unroll
  for (int i = 0; i < 2; ++i)
#pragma unroll
    for (int j = 0; j < 4; ++j) { aS[i][j] = (v8f)(0.0f); aT[i][j] = (v8f)(0.0f); }

  const int rsel = lane & 15;
  const int koff = (lane >> 4) * 8;

  for (int k0 = 0; k0 < K; k0 += BK) {
    __syncthreads();
    stage_a(A, (size_t)lda, blockM, k0, tid, As);
    stage_b(Ws, ldb, blockN, k0, tid, BsS);
    stage_b(Wt, ldb, blockN, k0, tid, BsT);
    if (k0 + BK < K)
      __builtin_prefetch(A + (blockM + (tid >> 1)) * (size_t)lda + k0 + BK, 0, 0);
    __syncthreads();

    v16bf af[2];
#pragma unroll
    for (int mt = 0; mt < 2; ++mt)
      af[mt] = load_frag(As, waveM * 32 + mt * 16 + rsel, koff);
#pragma unroll
    for (int nt = 0; nt < 4; ++nt) {
      const int col = waveN * 64 + nt * 16 + rsel;
      v16bf bS = load_frag(BsS, col, koff);
      v16bf bT = load_frag(BsT, col, koff);
#pragma unroll
      for (int mt = 0; mt < 2; ++mt) {
        aS[mt][nt] = __builtin_amdgcn_wmma_f32_16x16x32_bf16(
            false, af[mt], false, bS, (short)0, aS[mt][nt], false, false);
        aT[mt][nt] = __builtin_amdgcn_wmma_f32_16x16x32_bf16(
            false, af[mt], false, bT, (short)0, aT[mt][nt], false, false);
      }
    }
  }

  const int mlane = (lane >> 4) << 3;
  const int nlane = lane & 15;
#pragma unroll
  for (int mt = 0; mt < 2; ++mt)
#pragma unroll
    for (int nt = 0; nt < 4; ++nt)
#pragma unroll
      for (int r = 0; r < 8; ++r) {
        const size_t m = blockM + waveM * 32 + mt * 16 + r + mlane;
        const int n = blockN + waveN * 64 + nt * 16 + nlane;
        float vs = aS[mt][nt][r] + bs[n];
        float vt = aT[mt][nt][r] + bt[n];
        S[m * (size_t)ldc + n] = 1.0f / (1.0f + expf(-vs));
        T[m * (size_t)ldc + n] = tanhf(vt);
      }
}

// ------------------------------------------------- graph preprocessing ----
__global__ void gcn_degree_kernel(const int* __restrict__ ei, int* __restrict__ cnt,
                                  float* __restrict__ dinv) {
  const int n = blockIdx.x * blockDim.x + threadIdx.x;
  if (n >= NNODE) return;
  const int* dst = ei + NEDGE;
  int c = 0;
  for (int e = 0; e < NEDGE; ++e) c += (dst[e] == n);
  cnt[n] = c + 1;  // + self loop
  dinv[n] = rsqrtf(fmaxf((float)(c + 1), 1.0f));
}

__global__ void gcn_scan_kernel(const int* __restrict__ cnt, int* __restrict__ off) {
  if (blockIdx.x == 0 && threadIdx.x == 0) {
    int a = 0;
    for (int n = 0; n < NNODE; ++n) { off[n] = a; a += cnt[n]; }
    off[NNODE] = a;
  }
}

__global__ void gcn_fill_kernel(const int* __restrict__ ei, const int* __restrict__ off,
                                const float* __restrict__ dinv,
                                int* __restrict__ srcs, float* __restrict__ nrm) {
  const int n = blockIdx.x * blockDim.x + threadIdx.x;
  if (n >= NNODE) return;
  const int* s = ei;
  const int* d = ei + NEDGE;
  int p = off[n];
  const float dn = dinv[n];
  for (int e = 0; e < NEDGE; ++e) {
    if (d[e] == n) { srcs[p] = s[e]; nrm[p] = dn * dinv[s[e]]; ++p; }
  }
  srcs[p] = n;  // self loop last
  nrm[p]  = dn * dn;
}

// ----------------------------------------------- GCN aggregation + tanh ----
__global__ __launch_bounds__(256) void gcn_agg_kernel(
    const float* __restrict__ hw, const int* __restrict__ csr_off,
    const int* __restrict__ csr_src, const float* __restrict__ csr_nrm,
    const float* __restrict__ bias, float* __restrict__ out, int out_stride) {
  const int n = blockIdx.x, b = blockIdx.y, c = threadIdx.x;
  const int beg = csr_off[n], end = csr_off[n + 1];
  const float* base = hw + (size_t)b * NNODE * NHID;
  float sum = 0.0f;
  for (int i = beg; i < end; ++i)
    sum += csr_nrm[i] * base[(size_t)csr_src[i] * NHID + c];
  out[((size_t)b * NNODE + n) * (size_t)out_stride + c] = tanhf(sum + bias[c]);
}

// ------------------------------------------------------------ misc ops ----
__global__ void copy_x_kernel(const float* __restrict__ x, float* __restrict__ cat) {
  const size_t i = (size_t)blockIdx.x * blockDim.x + threadIdx.x;
  if (i >= (size_t)MROWS * FIN) return;
  const size_t row = i >> 7;
  const int c = (int)(i & (FIN - 1));
  cat[row * CATC + c] = x[i];
}

// b12[j] = sum_k b1[k] * W2[k,j]  (fused bias through the conv weight)
__global__ __launch_bounds__(256) void bias_fuse_kernel(
    const float* __restrict__ b1, const float* __restrict__ W2,
    float* __restrict__ b12) {
  const int j = threadIdx.x;
  float a = 0.0f;
  for (int k = 0; k < NHID; ++k) a += b1[k] * W2[k * NHID + j];
  b12[j] = a;
}

__global__ __launch_bounds__(256) void pool_kernel(
    const float* __restrict__ S, const float* __restrict__ T,
    float* __restrict__ emb_ws, float* __restrict__ emb_out) {
  const int b = blockIdx.x, c = threadIdx.x;
  const float* s = S + (size_t)b * NNODE * NHID + c;
  const float* t = T + (size_t)b * NNODE * NHID + c;
  float acc = 0.0f;
  for (int n = 0; n < NNODE; ++n)
    acc += s[(size_t)n * NHID] * t[(size_t)n * NHID];
  const float e = tanhf(acc);
  emb_ws[b * NHID + c] = e;
  emb_out[b * NHID + c] = e;
}

__global__ __launch_bounds__(256) void f1_kernel(const float* __restrict__ emb,
                                                 const float* __restrict__ W,
                                                 const float* __restrict__ bias,
                                                 float* __restrict__ h) {
  const int b = blockIdx.x, j = threadIdx.x;
  float acc = bias[j];
  for (int k = 0; k < NHID; ++k) acc += emb[b * NHID + k] * W[k * NHID + j];
  h[b * NHID + j] = fmaxf(acc, 0.0f);
}

__global__ void f2_kernel(const float* __restrict__ h, const float* __restrict__ W,
                          const float* __restrict__ bias, float* __restrict__ preds) {
  const int b = blockIdx.x, j = threadIdx.x;
  if (j >= 2) return;
  float acc = bias[j];
  for (int k = 0; k < NHID; ++k) acc += h[b * NHID + k] * W[k * 2 + j];
  if (j == 0) acc = 1.0f / (1.0f + expf(-acc));
  preds[b * 2 + j] = acc;
}

// ------------------------------------------------------------- launch -----
extern "C" void kernel_launch(void* const* d_in, const int* in_sizes, int n_in,
                              void* d_out, int out_size, void* d_ws, size_t ws_size,
                              hipStream_t stream) {
  const float* x       = (const float*)d_in[0];
  const int*   ei      = (const int*)d_in[1];
  const float* lin0_W  = (const float*)d_in[2];
  const float* lin0_b  = (const float*)d_in[3];
  const float* lin_Ws  = (const float*)d_in[4];   // [2,256,256]
  const float* lin_bs  = (const float*)d_in[5];   // [2,256]
  const float* conv_Ws = (const float*)d_in[6];   // [3,256,256]
  const float* conv_bs = (const float*)d_in[7];   // [3,256]
  const float* sig_W   = (const float*)d_in[8];
  const float* sig_b   = (const float*)d_in[9];
  const float* tanh_W  = (const float*)d_in[10];
  const float* tanh_b  = (const float*)d_in[11];
  const float* f1_W    = (const float*)d_in[12];
  const float* f1_b    = (const float*)d_in[13];
  const float* f2_W    = (const float*)d_in[14];
  const float* f2_b    = (const float*)d_in[15];
  float* out = (float*)d_out;  // [256*2 preds][256*256 embeds]

  size_t off = 0;
  auto carve = [&](size_t bytes) -> char* {
    char* p = (char*)d_ws + off;
    off += (bytes + 255) & ~(size_t)255;
    return p;
  };
  float* t0   = (float*)carve((size_t)MROWS * NHID * 4);  // S
  float* t1   = (float*)carve((size_t)MROWS * NHID * 4);  // layer out / T
  float* cat  = (float*)carve((size_t)MROWS * CATC * 4);
  float* w12  = (float*)carve(3 * NHID * NHID * 4);       // fused W1@W2 per layer
  float* b12  = (float*)carve(3 * NHID * 4);
  int*   cnt  = (int*)carve((NNODE + 1) * 4);
  int*   offs = (int*)carve((NNODE + 1) * 4);
  float* dinv = (float*)carve(NNODE * 4);
  int*   csrc = (int*)carve((NEDGE + NNODE) * 4);
  float* cnrm = (float*)carve((NEDGE + NNODE) * 4);
  float* emb  = (float*)carve(BATCH * NHID * 4);
  float* hbuf = (float*)carve(BATCH * NHID * 4);

  // 1) graph preprocessing (deterministic CSR, no atomics)
  gcn_degree_kernel<<<dim3(NNODE / 256), 256, 0, stream>>>(ei, cnt, dinv);
  gcn_scan_kernel<<<1, 1, 0, stream>>>(cnt, offs);
  gcn_fill_kernel<<<dim3(NNODE / 256), 256, 0, stream>>>(ei, offs, dinv, csrc, cnrm);

  // 2) x -> cat[:, :, 0:128]
  {
    const size_t tot = (size_t)MROWS * FIN;
    copy_x_kernel<<<dim3((unsigned)((tot + 255) / 256)), 256, 0, stream>>>(x, cat);
  }

  // 3) fold each layer's linear into its conv:  W12 = W1@W2,  b12 = b1@W2
  wmma_gemm_kernel<0><<<dim3(1, 2), 256, 0, stream>>>(
      lin0_W, NHID, conv_Ws, NHID, nullptr, w12, NHID, NHID);               // 128x256
  wmma_gemm_kernel<0><<<dim3(2, 2), 256, 0, stream>>>(
      lin_Ws, NHID, conv_Ws + NHID * NHID, NHID, nullptr, w12 + NHID * NHID, NHID, NHID);
  wmma_gemm_kernel<0><<<dim3(2, 2), 256, 0, stream>>>(
      lin_Ws + NHID * NHID, NHID, conv_Ws + 2 * NHID * NHID, NHID, nullptr,
      w12 + 2 * NHID * NHID, NHID, NHID);
  bias_fuse_kernel<<<1, 256, 0, stream>>>(lin0_b, conv_Ws, b12);
  bias_fuse_kernel<<<1, 256, 0, stream>>>(lin_bs, conv_Ws + NHID * NHID, b12 + NHID);
  bias_fuse_kernel<<<1, 256, 0, stream>>>(lin_bs + NHID, conv_Ws + 2 * NHID * NHID,
                                          b12 + 2 * NHID);

  const dim3 gemm_grid(MROWS / BM, NHID / BN);  // 1024 x 2
  const dim3 agg_grid(NNODE, BATCH);

  // 4) layer 0:  t1 = x @ W12_0 + b12_0 ; aggregate+tanh -> cat[:,:,128:384]
  wmma_gemm_kernel<0><<<gemm_grid, 256, 0, stream>>>(x, FIN, w12, NHID, b12, t1, NHID, FIN);
  gcn_agg_kernel<<<agg_grid, 256, 0, stream>>>(t1, offs, csrc, cnrm, conv_bs, cat + FIN, CATC);

  // 5) layer 1
  wmma_gemm_kernel<0><<<gemm_grid, 256, 0, stream>>>(
      cat + FIN, CATC, w12 + NHID * NHID, NHID, b12 + NHID, t1, NHID, NHID);
  gcn_agg_kernel<<<agg_grid, 256, 0, stream>>>(t1, offs, csrc, cnrm, conv_bs + NHID,
                                               cat + FIN + NHID, CATC);

  // 6) layer 2
  wmma_gemm_kernel<0><<<gemm_grid, 256, 0, stream>>>(
      cat + FIN + NHID, CATC, w12 + 2 * NHID * NHID, NHID, b12 + 2 * NHID, t1, NHID, NHID);
  gcn_agg_kernel<<<agg_grid, 256, 0, stream>>>(t1, offs, csrc, cnrm, conv_bs + 2 * NHID,
                                               cat + FIN + 2 * NHID, CATC);

  // 7) fused gated readout: S = sigmoid(cat@sigW+b) -> t0, T = tanh(cat@tanhW+b) -> t1
  wmma_gemm_dual_kernel<<<gemm_grid, 256, 0, stream>>>(
      cat, CATC, sig_W, tanh_W, NHID, sig_b, tanh_b, t0, t1, NHID, CATC);

  // 8) pooled = sum_n S*T ; embeds = tanh(pooled) -> d_out[512:]
  pool_kernel<<<dim3(BATCH), 256, 0, stream>>>(t0, t1, emb, out + BATCH * 2);

  // 9) head
  f1_kernel<<<dim3(BATCH), 256, 0, stream>>>(emb, f1_W, f1_b, hbuf);
  f2_kernel<<<dim3(BATCH), 64, 0, stream>>>(hbuf, f2_W, f2_b, out);
}
